// VQSendRecv_40312563040820
// MI455X (gfx1250) — compile-verified, matrix operand order checked by reading, not run on hardware
//
#include <hip/hip_runtime.h>
#include <hip/hip_bf16.h>
#include <math.h>

// ---------------------------------------------------------------------------
// VQ-VAE forward for MI455X (gfx1250, wave32, WMMA).
// All GEMMs use v_wmma_f32_16x16x32_bf16 (f32 accum).  All reductions are
// deterministic (fixed-order loops + LDS trees; no float atomics).
// ---------------------------------------------------------------------------

typedef __attribute__((ext_vector_type(16))) __bf16 v16bf;
typedef __attribute__((ext_vector_type(8)))  float  v8f;

#define NROWS   16384
#define DIN     1024
#define DEMB    256
#define KCODES  1024
#define ALPHA   0.01f
#define EPSV    1e-5f

// ---- workspace byte offsets -----------------------------------------------
#define WS_Z        0u          // 16384*256 f32   = 16777216 B
#define WS_CODES    16777216u   // 16384 i32       = 65536 B
#define WS_WSENDT   16842752u   // 256*1024 bf16   = 524288 B
#define WS_CB       17367040u   // 1024*256 bf16   = 524288 B
#define WS_WRECVT   17891328u   // 1024*256 bf16   = 524288 B
#define WS_SQR      18415616u   // 1024 f32
#define WS_COUNTS   18419712u   // 1024 i32
#define WS_ESN      18423808u   // 1024 f32
#define WS_SIZES    18427904u   // 1024 f32
#define WS_PART     18432000u   // 256 f32

// ---- d_out float-element offsets (tuple flattened in return order) --------
#define O_X       0u
#define O_CODES   16777216u
#define O_EMB     16793600u
#define O_COMMIT  16793601u
#define O_ENT     16793602u
#define O_EMAV    16793603u
#define O_EMAS    17055747u
#define O_W       17056771u

// ---------------------------------------------------------------------------
// Fragment loaders.  CDNA5 wave32 16-bit A layout (CDNA5 ISA):
// lane L: row m = L%16, half h = L/16; element j maps to
//   K = 16*(j/8) + 8*h + (j%8)
// so per lane we load 8 contiguous elems at (kb + 8h) and 8 at (kb + 8h + 16).
// B fragments use the identical layout when loaded from a column-major
// (i.e. pre-transposed row-major) copy of B.
// ---------------------------------------------------------------------------
__device__ __forceinline__ v16bf load_frag_bf16(const __bf16* __restrict__ rowp,
                                                int kb8h) {
  union { v16bf v; uint4 q[2]; } u;
  u.q[0] = *(const uint4*)(rowp + kb8h);
  u.q[1] = *(const uint4*)(rowp + kb8h + 16);
  return u.v;
}

struct f32frag { float f[16]; };

__device__ __forceinline__ f32frag load_raw_f32(const float* __restrict__ rowp,
                                                int kb8h) {
  f32frag r;
  *(float4*)(r.f + 0)  = *(const float4*)(rowp + kb8h);
  *(float4*)(r.f + 4)  = *(const float4*)(rowp + kb8h + 4);
  *(float4*)(r.f + 8)  = *(const float4*)(rowp + kb8h + 16);
  *(float4*)(r.f + 12) = *(const float4*)(rowp + kb8h + 20);
  return r;
}

__device__ __forceinline__ v16bf cvt_frag(const f32frag& r, float scale) {
  v16bf a;
#pragma unroll
  for (int j = 0; j < 16; ++j) a[j] = (__bf16)(r.f[j] * scale);
  return a;
}

// ---------------------------------------------------------------------------
// prep: generic transpose + f32->bf16.  dst[r*cols + c] = src[c*rows + r]
// ---------------------------------------------------------------------------
__global__ void k_transpose_bf16(__bf16* __restrict__ dst,
                                 const float* __restrict__ src,
                                 int rows_dst, int cols_dst) {
  int idx = blockIdx.x * 256 + threadIdx.x;
  int r = idx / cols_dst, c = idx - r * cols_dst;
  dst[idx] = (__bf16)src[(size_t)c * rows_dst + r];
}

// codebook -> bf16 copy + per-row squared norm
__global__ void k_cb_prep(const float* __restrict__ cb, __bf16* __restrict__ cb_bf,
                          float* __restrict__ sqr) {
  int k = blockIdx.x, t = threadIdx.x;
  float v = cb[(size_t)k * DEMB + t];
  cb_bf[(size_t)k * DEMB + t] = (__bf16)v;
  __shared__ float s[256];
  s[t] = v * v; __syncthreads();
  for (int w = 128; w > 0; w >>= 1) { if (t < w) s[t] += s[t + w]; __syncthreads(); }
  if (t == 0) sqr[k] = s[0];
}

// ---------------------------------------------------------------------------
// GEMM1: z[16384,256] = input[16384,1024] x W_send[1024,256] + b_send
// wave = 16x128 output strip, k-loop of 32, 8 WMMA per step.
// All 8 B fragments live in distinct registers so the 16 b128 loads issue as
// one clause and WMMAs retire against decreasing loadcnt (loads-in-order).
// ---------------------------------------------------------------------------
__global__ __launch_bounds__(256)
void k_gemm1(const float* __restrict__ inp, const __bf16* __restrict__ wsendT,
             const float* __restrict__ b_send, float* __restrict__ z) {
  int wid  = blockIdx.x * 8 + (threadIdx.x >> 5);   // 0..2047
  int mt   = wid >> 1;            // 0..1023
  int nb   = (wid & 1) * 128;     // 0 or 128
  int lane = threadIdx.x & 31, l16 = lane & 15, h = lane >> 4;
  int m0 = mt * 16;
  const float* arow = inp + (size_t)(m0 + l16) * DIN;
  v8f c[8] = {};
  for (int kb = 0; kb < DIN; kb += 32) {
    // issue A raw loads first, then all B loads, then convert, then compute
    f32frag araw = load_raw_f32(arow, kb + 8 * h);
    v16bf b[8];
#pragma unroll
    for (int s = 0; s < 8; ++s)
      b[s] = load_frag_bf16(wsendT + (size_t)(nb + 16 * s + l16) * DIN, kb + 8 * h);
    v16bf a = cvt_frag(araw, 1.0f);
#pragma unroll
    for (int s = 0; s < 8; ++s)
      c[s] = __builtin_amdgcn_wmma_f32_16x16x32_bf16(false, a, false, b[s],
                                                     (short)0, c[s], false, false);
  }
#pragma unroll
  for (int s = 0; s < 8; ++s) {
    int col = nb + 16 * s + l16;
    float bias = b_send[col];
#pragma unroll
    for (int r = 0; r < 8; ++r)
      z[(size_t)(m0 + r + 8 * h) * DEMB + col] = c[s][r] + bias;
  }
}

// ---------------------------------------------------------------------------
// Argmin: scores[n,k] = ||e_k||^2 - 2<z_n,e_k>.  A = (-2*z) cached as 8 frags,
// 64 column tiles x 8 k-steps = 512 WMMA/wave; C seeded with ||e_k||^2.
// Per-row argmin finished with a 16-lane shfl_xor butterfly.
// ---------------------------------------------------------------------------
__global__ __launch_bounds__(256)
void k_argmin(const float* __restrict__ z, const __bf16* __restrict__ cb,
              const float* __restrict__ sqr, int* __restrict__ codes_i,
              float* __restrict__ codes_f) {
  int wid  = blockIdx.x * 8 + (threadIdx.x >> 5);   // 0..1023
  int lane = threadIdx.x & 31, l16 = lane & 15, h = lane >> 4;
  int m0 = wid * 16;
  const float* arow = z + (size_t)(m0 + l16) * DEMB;
  v16bf a[8];
#pragma unroll
  for (int s = 0; s < 8; ++s)
    a[s] = cvt_frag(load_raw_f32(arow, 32 * s + 8 * h), -2.0f);

  float minv[8]; int mini[8];
#pragma unroll
  for (int r = 0; r < 8; ++r) { minv[r] = 3.4e38f; mini[r] = 0; }

  for (int kc = 0; kc < KCODES; kc += 16) {
    float sq = sqr[kc + l16];
    v16bf b[8];
#pragma unroll
    for (int s = 0; s < 8; ++s)
      b[s] = load_frag_bf16(cb + (size_t)(kc + l16) * DEMB, 32 * s + 8 * h);
    v8f c;
#pragma unroll
    for (int r = 0; r < 8; ++r) c[r] = sq;
#pragma unroll
    for (int s = 0; s < 8; ++s)
      c = __builtin_amdgcn_wmma_f32_16x16x32_bf16(false, a[s], false, b[s],
                                                  (short)0, c, false, false);
#pragma unroll
    for (int r = 0; r < 8; ++r)
      if (c[r] < minv[r]) { minv[r] = c[r]; mini[r] = kc + l16; }
  }
#pragma unroll
  for (int r = 0; r < 8; ++r) {
#pragma unroll
    for (int m = 1; m < 16; m <<= 1) {
      float ov = __shfl_xor(minv[r], m, 32);
      int   oi = __shfl_xor(mini[r], m, 32);
      if (ov < minv[r] || (ov == minv[r] && oi < mini[r])) { minv[r] = ov; mini[r] = oi; }
    }
    if (l16 == 0) {
      int row = m0 + r + 8 * h;
      codes_i[row] = mini[r];
      codes_f[row] = (float)mini[r];
    }
  }
}

// ---------------------------------------------------------------------------
// Deterministic segment-sum EMA: block = code k, thread t = column t.
// ---------------------------------------------------------------------------
__global__ void k_ema(const float* __restrict__ z, const int* __restrict__ codes,
                      const float* __restrict__ ema_vecs, const float* __restrict__ ema_size,
                      float* __restrict__ out_emav, float* __restrict__ out_emas,
                      int* __restrict__ counts, float* __restrict__ esn_ws) {
  int k = blockIdx.x, t = threadIdx.x;
  __shared__ int sc[256];
  float acc = 0.f; int cnt = 0;
  for (int n0 = 0; n0 < NROWS; n0 += 256) {
    sc[t] = codes[n0 + t];
    __syncthreads();
    for (int i = 0; i < 256; ++i) {
      if (sc[i] == k) { acc += z[(size_t)(n0 + i) * DEMB + t]; cnt++; }
    }
    __syncthreads();
  }
  float ev = ema_vecs[(size_t)k * DEMB + t];
  out_emav[(size_t)k * DEMB + t] = ev + ALPHA * (acc - ev);
  if (t == 0) {
    counts[k] = cnt;
    float es = ema_size[k];
    float esn = es + ALPHA * ((float)cnt - es);
    out_emas[k] = esn;
    esn_ws[k]   = esn;
  }
}

// loss partials: block b handles rows [64b, 64b+64), thread t = column t
__global__ void k_loss(const float* __restrict__ z, const int* __restrict__ codes,
                       const float* __restrict__ cb, float* __restrict__ partials) {
  int b = blockIdx.x, t = threadIdx.x;
  float acc = 0.f;
  for (int r = 0; r < 64; ++r) {
    int n = b * 64 + r;
    int c = codes[n];
    float d = cb[(size_t)c * DEMB + t] - z[(size_t)n * DEMB + t];
    acc += d * d;
  }
  __shared__ float s[256];
  s[t] = acc; __syncthreads();
  for (int w = 128; w > 0; w >>= 1) { if (t < w) s[t] += s[t + w]; __syncthreads(); }
  if (t == 0) partials[b] = s[0];
}

// scalars + smoothed sizes (single block)
__global__ void k_finalize(const int* __restrict__ counts, const float* __restrict__ esn,
                           const float* __restrict__ partials, float* __restrict__ sizes,
                           float* __restrict__ out) {
  int t = threadIdx.x;
  __shared__ float s1[256], s2[256], s3[256];
  __shared__ float coef;
  float sn = 0.f, se = 0.f;
  for (int k = t; k < KCODES; k += 256) {
    sn += esn[k];
    float p = (float)counts[k] * (1.0f / (float)NROWS);
    se += (p > 0.f) ? -p * log2f(p) : 0.f;
  }
  s1[t] = sn; s2[t] = se; s3[t] = partials[t];
  __syncthreads();
  for (int w = 128; w > 0; w >>= 1) {
    if (t < w) { s1[t] += s1[t + w]; s2[t] += s2[t + w]; s3[t] += s3[t + w]; }
    __syncthreads();
  }
  if (t == 0) {
    float n = s1[0];
    coef = n / (n + (float)KCODES * EPSV);
    out[O_EMB]    = s3[0];
    out[O_COMMIT] = s3[0];
    out[O_ENT]    = s2[0];
  }
  __syncthreads();
  for (int k = t; k < KCODES; k += 256) sizes[k] = coef * (esn[k] + EPSV);
}

__global__ void k_weight(const float* __restrict__ emav, const float* __restrict__ sizes,
                         float* __restrict__ w) {
  int idx = blockIdx.x * 256 + threadIdx.x;
  w[idx] = emav[idx] / sizes[idx >> 8];
}

// ---------------------------------------------------------------------------
// GEMM3: x[16384,1024] = codebook[codes] x W_recv + b_recv
// wave = 16x64 strip; A (gathered codebook rows) cached as 8 frags; 32 WMMA
// ---------------------------------------------------------------------------
__global__ __launch_bounds__(256)
void k_gemm3(const int* __restrict__ codes_i, const __bf16* __restrict__ cb,
             const __bf16* __restrict__ wrecvT, const float* __restrict__ b_recv,
             float* __restrict__ x) {
  int wid  = blockIdx.x * 8 + (threadIdx.x >> 5);
  int mt   = wid >> 4;            // 0..1023
  int nt   = wid & 15;            // 0..15
  int lane = threadIdx.x & 31, l16 = lane & 15, h = lane >> 4;
  int m0 = mt * 16, nb = nt * 64;
  int code = codes_i[m0 + l16];
  const __bf16* arow = cb + (size_t)code * DEMB;
  v16bf a[8];
#pragma unroll
  for (int s = 0; s < 8; ++s) a[s] = load_frag_bf16(arow, 32 * s + 8 * h);
#pragma unroll
  for (int tsub = 0; tsub < 4; ++tsub) {
    int colb = nb + 16 * tsub;
    v16bf b[8];
#pragma unroll
    for (int s = 0; s < 8; ++s)
      b[s] = load_frag_bf16(wrecvT + (size_t)(colb + l16) * DEMB, 32 * s + 8 * h);
    v8f c = {};
#pragma unroll
    for (int s = 0; s < 8; ++s)
      c = __builtin_amdgcn_wmma_f32_16x16x32_bf16(false, a[s], false, b[s],
                                                  (short)0, c, false, false);
    int col = colb + l16;
    float bias = b_recv[col];
#pragma unroll
    for (int r = 0; r < 8; ++r)
      x[(size_t)(m0 + r + 8 * h) * DIN + col] = c[r] + bias;
  }
}

// ---------------------------------------------------------------------------
extern "C" void kernel_launch(void* const* d_in, const int* in_sizes, int n_in,
                              void* d_out, int out_size, void* d_ws, size_t ws_size,
                              hipStream_t stream) {
  const float* inp      = (const float*)d_in[0];
  const float* W_send   = (const float*)d_in[1];
  const float* b_send   = (const float*)d_in[2];
  const float* W_recv   = (const float*)d_in[3];
  const float* b_recv   = (const float*)d_in[4];
  const float* codebook = (const float*)d_in[5];
  const float* ema_vecs = (const float*)d_in[6];
  const float* ema_size = (const float*)d_in[7];

  char* ws = (char*)d_ws;
  float*  z       = (float*)(ws + WS_Z);
  int*    codes_i = (int*)(ws + WS_CODES);
  __bf16* wsendT  = (__bf16*)(ws + WS_WSENDT);
  __bf16* cb_bf   = (__bf16*)(ws + WS_CB);
  __bf16* wrecvT  = (__bf16*)(ws + WS_WRECVT);
  float*  sqr     = (float*)(ws + WS_SQR);
  int*    counts  = (int*)(ws + WS_COUNTS);
  float*  esn     = (float*)(ws + WS_ESN);
  float*  sizes   = (float*)(ws + WS_SIZES);
  float*  part    = (float*)(ws + WS_PART);

  float* out = (float*)d_out;

  // prep: transposed bf16 weights + bf16 codebook + ||e_k||^2
  k_transpose_bf16<<<1024, 256, 0, stream>>>(wsendT, W_send, DEMB, DIN);
  k_transpose_bf16<<<1024, 256, 0, stream>>>(wrecvT, W_recv, DIN, DEMB);
  k_cb_prep<<<KCODES, 256, 0, stream>>>(codebook, cb_bf, sqr);

  // z = input @ W_send + b_send
  k_gemm1<<<256, 256, 0, stream>>>(inp, wsendT, b_send, z);

  // nearest-neighbor codes
  k_argmin<<<128, 256, 0, stream>>>(z, cb_bf, sqr, codes_i, out + O_CODES);

  // EMA segment-sum + losses
  k_ema<<<KCODES, 256, 0, stream>>>(z, codes_i, ema_vecs, ema_size,
                                    out + O_EMAV, out + O_EMAS, counts, esn);
  k_loss<<<256, 256, 0, stream>>>(z, codes_i, codebook, part);
  k_finalize<<<1, 256, 0, stream>>>(counts, esn, part, sizes, out);
  k_weight<<<1024, 256, 0, stream>>>(out + O_EMAV, sizes, out + O_W);

  // x = vectors @ W_recv + b_recv  (emb_out == vectors numerically)
  k_gemm3<<<2048, 256, 0, stream>>>(codes_i, cb_bf, wrecvT, b_recv, out + O_X);
}
